// ABCLayer_17016660427527
// MI455X (gfx1250) — compile-verified, matrix operand order checked by reading, not run on hardware
//
#include <hip/hip_runtime.h>
#include <math.h>

typedef _Float16 half_t;
typedef __attribute__((ext_vector_type(16))) _Float16 v16h;
typedef __attribute__((ext_vector_type(8)))  _Float16 v8h;
typedef __attribute__((ext_vector_type(8)))  float    v8f;

#define K_ELEMS 294912          // 3*3*128*256
#define KDIM    1152            // 9*128
#define BATCH   8
#define HH      32
#define WWID    32
#define CIN     128
#define COUT    256
#define HP      34
#define WP      34
#define MBASES  5
#define NBR     3

// workspace layout (float indices / byte offsets)
#define WS_STAT_SUM  0        // 576 floats
#define WS_STAT_SQ   576      // 576 floats
#define WS_MEAN      1200
#define WS_STD       1201
#define WS_GRAM      1216     // 288*20 floats
#define WS_ALPHA     7040     // 5 floats
#define WS_HALF_BYTE 32768    // byte offset: wt (294912 halves, K-major) then xp (1183744 halves)

__device__ __forceinline__ float sgnf(float v) {
    return (v > 0.f) ? 1.f : ((v < 0.f) ? -1.f : 0.f);
}

// ---------------- X_eff: padded (8,34,34,128) f16, zero halo -------------
__global__ void k_xeff(const float* __restrict__ x, const float* __restrict__ shiftPara,
                       const float* __restrict__ beta, half_t* __restrict__ xp) {
    int idx = blockIdx.x * 256 + threadIdx.x;          // 0 .. 1183743 (exact)
    int ci = idx & (CIN - 1);
    int p  = idx >> 7;
    int xq = p % WP; p /= WP;
    int yq = p % HP;
    int b  = p / HP;
    half_t r = (half_t)0.f;
    if (yq >= 1 && yq <= HH && xq >= 1 && xq <= WWID) {
        float xv = x[((b * HH + (yq - 1)) * WWID + (xq - 1)) * CIN + ci];
        float s = 0.f;
#pragma unroll
        for (int n = 0; n < NBR; n++) {
            float t = xv + shiftPara[n];
            t = fminf(fmaxf(t, 0.f), 1.f) - 0.5f;
            s += beta[n] * sgnf(t);
        }
        r = (half_t)s;
    }
    xp[idx] = r;
}

// ---------------- weight mean/var, stage 1 (576 blocks) ------------------
__global__ void k_stats1(const float* __restrict__ w, float* wsf) {
    __shared__ float s1[256], s2[256];
    int tid = threadIdx.x;
    float a = 0.f, b = 0.f;
    int base = blockIdx.x * 512 + tid;
#pragma unroll
    for (int e = 0; e < 2; e++) { float v = w[base + 256 * e]; a += v; b += v * v; }
    s1[tid] = a; s2[tid] = b; __syncthreads();
    for (int st = 128; st > 0; st >>= 1) {
        if (tid < st) { s1[tid] += s1[tid + st]; s2[tid] += s2[tid + st]; }
        __syncthreads();
    }
    if (tid == 0) { wsf[WS_STAT_SUM + blockIdx.x] = s1[0]; wsf[WS_STAT_SQ + blockIdx.x] = s2[0]; }
}

// ---------------- stage 2: mean, std --------------------------------------
__global__ void k_stats2(float* wsf) {
    __shared__ float s1[256], s2[256];
    int tid = threadIdx.x;
    float a = 0.f, b = 0.f;
    for (int i = tid; i < 576; i += 256) { a += wsf[WS_STAT_SUM + i]; b += wsf[WS_STAT_SQ + i]; }
    s1[tid] = a; s2[tid] = b; __syncthreads();
    for (int st = 128; st > 0; st >>= 1) {
        if (tid < st) { s1[tid] += s1[tid + st]; s2[tid] += s2[tid + st]; }
        __syncthreads();
    }
    if (tid == 0) {
        float mean = s1[0] / (float)K_ELEMS;
        float var  = s2[0] / (float)K_ELEMS - mean * mean;
        wsf[WS_MEAN] = mean;
        wsf[WS_STD]  = sqrtf(fmaxf(var, 0.f));
    }
}

// ---------------- Gram = B B^T (15 uniq) and c = B w, stage 1 (288 blocks)
__global__ void k_gram1(const float* __restrict__ w, float* wsf) {
    __shared__ float red[256][21];
    int tid = threadIdx.x;
    float mean = wsf[WS_MEAN], stdv = wsf[WS_STD];
    float sh[MBASES];
#pragma unroll
    for (int m = 0; m < MBASES; m++) sh[m] = (-1.f + 0.5f * (float)m) * stdv;
    float g[15]; float c[5];
#pragma unroll
    for (int q = 0; q < 15; q++) g[q] = 0.f;
#pragma unroll
    for (int m = 0; m < 5; m++) c[m] = 0.f;
    int base = blockIdx.x * 1024 + tid;
#pragma unroll
    for (int e = 0; e < 4; e++) {
        float wv = w[base + 256 * e];
        float s[MBASES];
#pragma unroll
        for (int m = 0; m < MBASES; m++) s[m] = sgnf(wv - mean + sh[m]);
        int p = 0;
#pragma unroll
        for (int i = 0; i < MBASES; i++)
#pragma unroll
            for (int j = i; j < MBASES; j++) { g[p] += s[i] * s[j]; p++; }
#pragma unroll
        for (int m = 0; m < MBASES; m++) c[m] += s[m] * wv;
    }
#pragma unroll
    for (int q = 0; q < 15; q++) red[tid][q] = g[q];
#pragma unroll
    for (int m = 0; m < 5; m++) red[tid][15 + m] = c[m];
    __syncthreads();
    for (int st = 128; st > 0; st >>= 1) {
        if (tid < st) {
#pragma unroll
            for (int q = 0; q < 20; q++) red[tid][q] += red[tid + st][q];
        }
        __syncthreads();
    }
    if (tid == 0) {
#pragma unroll
        for (int q = 0; q < 20; q++) wsf[WS_GRAM + blockIdx.x * 20 + q] = red[0][q];
    }
}

// ---------------- reduce Gram, then 500-step GD on 5x5 --------------------
__global__ void k_alphas(float* wsf, const float* __restrict__ a0) {
    __shared__ float tot[20];
    int tid = threadIdx.x;
    if (tid < 20) {
        float s = 0.f;
        for (int b = 0; b < 288; b++) s += wsf[WS_GRAM + b * 20 + tid];
        tot[tid] = s;
    }
    __syncthreads();
    if (tid == 0) {
        float G[5][5], c[5], a[5];
        int p = 0;
        for (int i = 0; i < 5; i++)
            for (int j = i; j < 5; j++) { G[i][j] = tot[p]; G[j][i] = tot[p]; p++; }
        for (int m = 0; m < 5; m++) { c[m] = tot[15 + m]; a[m] = a0[m]; }
        const float invK = 1.f / (float)K_ELEMS;
        for (int it = 0; it < 500; it++) {
            float grd[5];
#pragma unroll
            for (int m = 0; m < 5; m++) {
                float s = 0.f;
#pragma unroll
                for (int j = 0; j < 5; j++) s += G[m][j] * a[j];
                grd[m] = (s - c[m]) * invK;
            }
#pragma unroll
            for (int m = 0; m < 5; m++) a[m] -= 0.01f * grd[m];
        }
        for (int m = 0; m < 5; m++) wsf[WS_ALPHA + m] = a[m];
    }
}

// ------ W_eff, stored K-major: wt[cout][k], k = (kh*3+kw)*128+ci (f16) ----
__global__ void k_weff(const float* __restrict__ w, const float* __restrict__ wsf,
                       half_t* __restrict__ wt) {
    int idx = blockIdx.x * 256 + threadIdx.x;          // 0 .. 294911 (exact)
    float mean = wsf[WS_MEAN], stdv = wsf[WS_STD];
    float wv = w[idx];
    float s = 0.f;
#pragma unroll
    for (int m = 0; m < MBASES; m++) {
        float shm = (-1.f + 0.5f * (float)m) * stdv;
        s += wsf[WS_ALPHA + m] * sgnf(wv - mean + shm);
    }
    int cout = idx & (COUT - 1);
    int k    = idx >> 8;
    wt[cout * KDIM + k] = (half_t)s;
}

// ---------------- register-resident implicit-GEMM conv via WMMA -----------
// M = 8192 pixels, N = 256 couts, K = 1152 (k-index = khw*128+ci).
// Wave tile: 32 rows x 64 cols = 8 WMMAs / K-step. No LDS, no barriers:
// both operands are K-major so each CDNA5 f16 fragment is exactly two
// 16B-aligned b128 global loads per lane (halves [koff,koff+8) and
// [16+koff,16+koff+8) of the 32-k chunk), served out of L2/WGP$.
// All 12 loads of a K-step are materialized before the 8 WMMAs so the
// backend can clause them and use partial loadcnt waits.
__global__ __launch_bounds__(128) void k_conv(const half_t* __restrict__ xp,
                                              const half_t* __restrict__ wt,
                                              float* __restrict__ out) {
    int tid  = threadIdx.x;
    int wave = tid >> 5;
    int lane = tid & 31;
    int lrow = lane & 15;
    int koff = (lane >= 16) ? 8 : 0;
    int m0 = blockIdx.x * 128;          // 4 waves x 32 rows
    int n0 = blockIdx.y * 64;

    // per-lane pixel bases for the two 16-row groups (padded, (dy,dx)=(0,0))
    int p00[2];
#pragma unroll
    for (int g = 0; g < 2; g++) {
        int m = m0 + wave * 32 + g * 16 + lrow;
        int b = m >> 10; int rem = m & 1023;
        int y = rem >> 5; int xc = rem & 31;
        p00[g] = ((b * HP + y) * WP + xc) * CIN;
    }
    // per-lane B row bases (K-major weights)
    int bb[4];
#pragma unroll
    for (int j = 0; j < 4; j++) bb[j] = (n0 + j * 16 + lrow) * KDIM;

    v8f acc[2][4];
#pragma unroll
    for (int g = 0; g < 2; g++)
#pragma unroll
        for (int j = 0; j < 4; j++) acc[g][j] = (v8f){};

    for (int t = 0; t < 36; t++) {
        int k0  = t * 32;
        int khw = k0 >> 7;              // constant within a 32-chunk (32 | 128)
        int ci0 = k0 & 127;
        int dy  = khw / 3, dx = khw % 3;
        int aoff = (dy * WP + dx) * CIN + ci0 + koff;

        // -------- issue every load of this K-step up front --------
        v16h af[2];
#pragma unroll
        for (int g = 0; g < 2; g++) {
            v8h lo = *(const v8h*)(xp + p00[g] + aoff);
            v8h hi = *(const v8h*)(xp + p00[g] + aoff + 16);
            af[g] = __builtin_shufflevector(lo, hi,
                      0, 1, 2, 3, 4, 5, 6, 7, 8, 9, 10, 11, 12, 13, 14, 15);
        }
        v16h bf[4];
#pragma unroll
        for (int j = 0; j < 4; j++) {
            v8h lo = *(const v8h*)(wt + bb[j] + k0 + koff);
            v8h hi = *(const v8h*)(wt + bb[j] + k0 + koff + 16);
            bf[j] = __builtin_shufflevector(lo, hi,
                      0, 1, 2, 3, 4, 5, 6, 7, 8, 9, 10, 11, 12, 13, 14, 15);
        }
        if (t + 1 < 36) {
            int k1 = k0 + 32; int kh1 = k1 >> 7;
            int aoff1 = ((kh1 / 3) * WP + (kh1 % 3)) * CIN + (k1 & 127) + koff;
            __builtin_prefetch(xp + p00[0] + aoff1, 0, 3);   // near-cache scope
        }
        // -------- 8 WMMAs --------
#pragma unroll
        for (int j = 0; j < 4; j++)
#pragma unroll
            for (int g = 0; g < 2; g++) {
                acc[g][j] = __builtin_amdgcn_wmma_f32_16x16x32_f16(
                    false, af[g], false, bf[j], (short)0, acc[g][j], false, false);
            }
    }

    // D layout: VGPR r -> M = r + 8*(lane>=16), N = lane&15
    int rowoff = (lane >= 16) ? 8 : 0;
#pragma unroll
    for (int g = 0; g < 2; g++) {
#pragma unroll
        for (int j = 0; j < 4; j++) {
#pragma unroll
            for (int r = 0; r < 8; r++) {
                int m = m0 + wave * 32 + g * 16 + rowoff + r;
                out[m * COUT + n0 + j * 16 + lrow] = acc[g][j][r];
            }
        }
    }
}

extern "C" void kernel_launch(void* const* d_in, const int* in_sizes, int n_in,
                              void* d_out, int out_size, void* d_ws, size_t ws_size,
                              hipStream_t stream) {
    (void)in_sizes; (void)n_in; (void)out_size; (void)ws_size;
    const float* x         = (const float*)d_in[0];   // (8,32,32,128)
    const float* weight    = (const float*)d_in[1];   // (3,3,128,256)
    const float* shiftPara = (const float*)d_in[2];   // (3,)
    const float* beta      = (const float*)d_in[3];   // (3,)
    const float* a0        = (const float*)d_in[4];   // (5,)
    float*  out = (float*)d_out;                      // (8,32,32,256)
    float*  wsf = (float*)d_ws;
    half_t* wt  = (half_t*)((char*)d_ws + WS_HALF_BYTE);   // 256 x 1152, K-major
    half_t* xp  = wt + K_ELEMS;                            // 8 x 34 x 34 x 128

    k_xeff  <<<4624, 256, 0, stream>>>(x, shiftPara, beta, xp);
    k_stats1<<< 576, 256, 0, stream>>>(weight, wsf);
    k_stats2<<<   1, 256, 0, stream>>>(wsf);
    k_gram1 <<< 288, 256, 0, stream>>>(weight, wsf);
    k_alphas<<<   1,  32, 0, stream>>>(wsf, a0);
    k_weff  <<<1152, 256, 0, stream>>>(weight, wsf, wt);
    k_conv  <<<dim3(64, 4), 128, 0, stream>>>(xp, wt, out);
}
